// VQVAE_52432960750061
// MI455X (gfx1250) — matrix operand order repaired and checked by reading, not compiled, not run on hardware
//
#include <hip/hip_runtime.h>
#include <hip/hip_bf16.h>
#include <math.h>

typedef __attribute__((ext_vector_type(16))) _Float16 v16h;
typedef __attribute__((ext_vector_type(8)))  _Float16 v8h;
typedef __attribute__((ext_vector_type(8)))  float    v8f;

#define K_EMB 4096
#define DIM   256
#define N_IN  32768
#define LDS_STRIDE 264       // 256 + 8 halves pad -> conflict-free 16-lane column reads
#define STAGE_HALVES (64 * LDS_STRIDE)
#define STAGE_BYTES  (STAGE_HALVES * 2)
#define KSPLIT 2             // codebook halves, one per block group
#define KHALF  (K_EMB / KSPLIT)
#define NSTAGE (KHALF / 64)

// ---------------------------------------------------------------------------
// Kernel 1: Wh = f16(W) (pre-converted codebook, 2 MB / L2-resident) and
//           wnorm[k] = sum_d W[k][d]^2
// ---------------------------------------------------------------------------
__global__ __launch_bounds__(256) void vq_prep(const float* __restrict__ W,
                                               _Float16* __restrict__ Wh,
                                               float* __restrict__ wnorm) {
    __shared__ float red[256];
    const int row = blockIdx.x;
    const int d   = threadIdx.x;
    const float v = W[(size_t)row * DIM + d];
    Wh[(size_t)row * DIM + d] = (_Float16)v;
    red[d] = v * v;
    __syncthreads();
    for (int s = 128; s > 0; s >>= 1) {
        if (d < s) red[d] += red[d + s];
        __syncthreads();
    }
    if (d == 0) wnorm[row] = red[0];
}

// ---------------------------------------------------------------------------
// Kernel 2: fused  score = ||w||^2 - 2 x.w  + running argmin, via WMMA f16.
// Block = 8 waves; wave owns 32 rows of X (two A register sets) so every
// staged B fragment feeds TWO wmmas; B fragment LDS reads double-buffered.
// W tiles staged f16 -> LDS with global_load_async_to_lds_b128 (ASYNCcnt),
// double-buffered at stage level: stage s+1 fills while stage s computes.
// The async ops' LDS addresses are derived from &tileW[...] so the compiler
// knows the asm writes the LDS allocation (prevents load folding).
// Grid = 128 row-groups x 2 codebook halves -> partial (score, idx) per row.
// ---------------------------------------------------------------------------
__global__ __launch_bounds__(256) void vq_argmin(const float* __restrict__ X,
                                                 const _Float16* __restrict__ Wh,
                                                 const float* __restrict__ wnorm,
                                                 float* __restrict__ pScore,
                                                 int* __restrict__ pIdx) {
    __shared__ __align__(32) _Float16 tileW[2 * STAGE_HALVES];

    const int tid  = threadIdx.x;
    const int wave = tid >> 5;
    const int lane = tid & 31;
    const int hi   = lane >> 4;     // half-wave select
    const int col  = lane & 15;
    const int rowGroup = blockIdx.x & 127;
    const int ksplit   = blockIdx.x >> 7;
    const int rowBase  = rowGroup * 256 + wave * 32;
    const int kbase    = ksplit * KHALF;

    // per-thread staging geometry: 8 chunks of 16B per thread per stage
    // i = tid + j*256 -> row = i>>5 (embedding in stage), chunk = i&31
    const int srow   = tid >> 5;          // rows srow, srow+8, ..., srow+56
    const int schunk = tid & 31;          // 16B chunk within the 512B row
    const unsigned sgoffBase = (unsigned)((srow * DIM + schunk * 8) * 2);
    // flat pointer to this thread's first LDS chunk; low 32 bits == DS address
    _Float16* const sldsBase = tileW + srow * LDS_STRIDE + schunk * 8;

    // ---- A fragments: 2 sets x 16 rows x 256 dims (16x32 A, contiguous k order, same as B) ----
    // lane group hi: elems 0..15 -> K = 32f + 16*hi + e
    v16h a0[8], a1[8];
    {
        const float* xr0 = X + (size_t)(rowBase + col) * DIM;
        const float* xr1 = X + (size_t)(rowBase + 16 + col) * DIM;
        #pragma unroll
        for (int f = 0; f < 8; ++f) {
            const int kb = f * 32 + hi * 16;
            v16h t0, t1;
            #pragma unroll
            for (int e = 0; e < 8; ++e) {
                t0[e]     = (_Float16)xr0[kb + e];
                t0[8 + e] = (_Float16)xr0[kb + 8 + e];
                t1[e]     = (_Float16)xr1[kb + e];
                t1[8 + e] = (_Float16)xr1[kb + 8 + e];
            }
            a0[f] = t0;
            a1[f] = t1;
        }
    }

    float best0[8], best1[8];
    int   bidx0[8], bidx1[8];
    #pragma unroll
    for (int r = 0; r < 8; ++r) {
        best0[r] = 3.0e38f; bidx0[r] = 0;
        best1[r] = 3.0e38f; bidx1[r] = 0;
    }

    // ---- async-fill of one 64-embedding stage into LDS buffer 'parity' ----
    auto issue_stage = [&](int stageIdx, int parity) {
        const unsigned long long gbase =
            (unsigned long long)(Wh + (size_t)(kbase + stageIdx * 64) * DIM);
        // DS byte address from the real LDS pointer (escapes into the asm)
        const unsigned lbase =
            (unsigned)(unsigned long long)(sldsBase + parity * STAGE_HALVES);
        #pragma unroll
        for (int j = 0; j < 8; ++j) {     // 8 rows per thread, stride 8 rows
            const unsigned goff = sgoffBase + (unsigned)(j * 8 * DIM * 2);
            const unsigned loff = lbase    + (unsigned)(j * 8 * LDS_STRIDE * 2);
            asm volatile("global_load_async_to_lds_b128 %0, %1, %2"
                         :: "v"(loff), "v"(goff), "s"(gbase) : "memory");
        }
    };

    // prologue: fill stage 0, wait, make visible
    issue_stage(0, 0);
    asm volatile("s_wait_asynccnt 0x0" ::: "memory");
    __syncthreads();

    for (int stage = 0; stage < NSTAGE; ++stage) {
        const int p = stage & 1;
        if (stage + 1 < NSTAGE)            // overlap next fill with this compute
            issue_stage(stage + 1, p ^ 1);

        const _Float16* bufBase = tileW + p * STAGE_HALVES;

        #pragma unroll
        for (int st = 0; st < 4; ++st) {
            const int nloc  = st * 16 + col;          // embedding within stage
            const int nglob = kbase + stage * 64 + nloc;
            // B layout (16-bit 32x16): lane col = N, per-lane K = 32f + 16*hi + e
            const _Float16* bp = bufBase + nloc * LDS_STRIDE + hi * 16;

            v8f c0 = {0.f, 0.f, 0.f, 0.f, 0.f, 0.f, 0.f, 0.f};
            v8f c1 = {0.f, 0.f, 0.f, 0.f, 0.f, 0.f, 0.f, 0.f};

            // double-buffered B fragment pipeline: frag f+1 loads in flight
            // while frag f feeds two wmmas (a0, a1 row sets)
            v8h blo = *(const v8h*)(bp);
            v8h bhi = *(const v8h*)(bp + 8);
            #pragma unroll
            for (int f = 0; f < 8; ++f) {
                v8h nlo = blo, nhi = bhi;
                if (f < 7) {
                    nlo = *(const v8h*)(bp + (f + 1) * 32);
                    nhi = *(const v8h*)(bp + (f + 1) * 32 + 8);
                }
                v16h b = __builtin_shufflevector(blo, bhi,
                          0,1,2,3,4,5,6,7,8,9,10,11,12,13,14,15);
                c0 = __builtin_amdgcn_wmma_f32_16x16x32_f16(
                        false, a0[f], false, b, (short)0, c0, false, false);
                c1 = __builtin_amdgcn_wmma_f32_16x16x32_f16(
                        false, a1[f], false, b, (short)0, c1, false, false);
                blo = nlo; bhi = nhi;
            }

            const float wn = wnorm[nglob];
            // D layout: vgpr r, this lane -> (row = base + r + 8*hi, col = nglob)
            #pragma unroll
            for (int r = 0; r < 8; ++r) {
                const float s0 = fmaf(-2.0f, c0[r], wn);
                if (s0 < best0[r]) { best0[r] = s0; bidx0[r] = nglob; }
                const float s1 = fmaf(-2.0f, c1[r], wn);
                if (s1 < best1[r]) { best1[r] = s1; bidx1[r] = nglob; }
            }
        }

        // next stage's async fill must be complete & visible before next iter
        asm volatile("s_wait_asynccnt 0x0" ::: "memory");
        __syncthreads();
    }

    // ---- reduce argmin across the 16 column-lanes of each half-wave ----
    #pragma unroll
    for (int off = 1; off < 16; off <<= 1) {
        #pragma unroll
        for (int r = 0; r < 8; ++r) {
            float ob = __shfl_xor(best0[r], off, 32);
            int   oi = __shfl_xor(bidx0[r], off, 32);
            if (ob < best0[r] || (ob == best0[r] && oi < bidx0[r])) {
                best0[r] = ob; bidx0[r] = oi;
            }
            ob = __shfl_xor(best1[r], off, 32);
            oi = __shfl_xor(bidx1[r], off, 32);
            if (ob < best1[r] || (ob == best1[r] && oi < bidx1[r])) {
                best1[r] = ob; bidx1[r] = oi;
            }
        }
    }
    if (col == 0) {
        const size_t base = (size_t)ksplit * N_IN;
        const int rb0 = rowBase + hi * 8;
        const int rb1 = rowBase + 16 + hi * 8;
        #pragma unroll
        for (int r = 0; r < 8; ++r) {
            pScore[base + rb0 + r] = best0[r];
            pIdx  [base + rb0 + r] = bidx0[r];
            pScore[base + rb1 + r] = best1[r];
            pIdx  [base + rb1 + r] = bidx1[r];
        }
    }
}

// ---------------------------------------------------------------------------
// Kernel 3: merge the two codebook-half partials (split-0 wins ties ->
// matches jnp.argmin first-hit since split-0 indices are smaller).
// ---------------------------------------------------------------------------
__global__ __launch_bounds__(256) void vq_merge(const float* __restrict__ pScore,
                                                const int* __restrict__ pIdx,
                                                int* __restrict__ bestIdx) {
    const int i = blockIdx.x * 256 + threadIdx.x;
    const float s0 = pScore[i];
    const float s1 = pScore[(size_t)N_IN + i];
    const int   i0 = pIdx[i];
    const int   i1 = pIdx[(size_t)N_IN + i];
    bestIdx[i] = (s1 < s0) ? i1 : i0;
}

// ---------------------------------------------------------------------------
// Kernel 4: gather quantized rows (full f32 W), per-row SSE partials.
// quantized_st == inputs + (quantized - inputs) == quantized (value-wise).
// ---------------------------------------------------------------------------
__global__ __launch_bounds__(256) void vq_gather(const float* __restrict__ X,
                                                 const float* __restrict__ W,
                                                 const int* __restrict__ bestIdx,
                                                 float* __restrict__ out,
                                                 float* __restrict__ partials) {
    __shared__ float red[256];
    const int n = blockIdx.x;
    const int d = threadIdx.x;
    const int k = bestIdx[n];                    // uniform -> scalar load
    const float q = W[(size_t)k * DIM + d];
    const float x = X[(size_t)n * DIM + d];
    out[1 + (size_t)n * DIM + d] = q;            // straight-through value
    const float diff = q - x;
    red[d] = diff * diff;
    __syncthreads();
    for (int s = 128; s > 0; s >>= 1) {
        if (d < s) red[d] += red[d + s];
        __syncthreads();
    }
    if (d == 0) partials[n] = red[0];
}

// ---------------------------------------------------------------------------
// Kernel 5: codebook usage counts (integer atomics -> deterministic)
// ---------------------------------------------------------------------------
__global__ __launch_bounds__(256) void vq_counts(const int* __restrict__ bestIdx,
                                                 int* __restrict__ counts) {
    const int i = blockIdx.x * blockDim.x + threadIdx.x;
    if (i < N_IN) atomicAdd(&counts[bestIdx[i]], 1);
}

// ---------------------------------------------------------------------------
// Kernel 6: loss = 1.25 * MSE ; perplexity = exp(-sum p*log(p+1e-10))
// ---------------------------------------------------------------------------
__global__ __launch_bounds__(256) void vq_finalize(const float* __restrict__ partials,
                                                   const int* __restrict__ counts,
                                                   float* __restrict__ out) {
    __shared__ float red[256];
    const int t = threadIdx.x;

    float s = 0.f;
    for (int i = t; i < N_IN; i += 256) s += partials[i];
    red[t] = s;
    __syncthreads();
    for (int st = 128; st > 0; st >>= 1) {
        if (t < st) red[t] += red[t + st];
        __syncthreads();
    }
    const float sse = red[0];
    __syncthreads();

    float h = 0.f;
    for (int i = t; i < K_EMB; i += 256) {
        const float p = (float)counts[i] / (float)N_IN;
        h += p * logf(p + 1e-10f);
    }
    red[t] = h;
    __syncthreads();
    for (int st = 128; st > 0; st >>= 1) {
        if (t < st) red[t] += red[t + st];
        __syncthreads();
    }
    if (t == 0) {
        const float mse = sse / (float)((size_t)N_IN * DIM);
        out[0] = 1.25f * mse;                       // q_loss + 0.25*e_loss, equal terms
        out[1 + (size_t)N_IN * DIM] = expf(-red[0]);
    }
}

// ---------------------------------------------------------------------------
extern "C" void kernel_launch(void* const* d_in, const int* in_sizes, int n_in,
                              void* d_out, int out_size, void* d_ws, size_t ws_size,
                              hipStream_t stream) {
    const float* X = (const float*)d_in[0];   // [N_IN, DIM]
    const float* W = (const float*)d_in[1];   // [K_EMB, DIM]
    float* out = (float*)d_out;               // [1 + N_IN*DIM + 1]

    char* ws = (char*)d_ws;
    _Float16* Wh     = (_Float16*)ws;                       // 2 MB  (f16 codebook)
    float* wnorm     = (float*)(ws + (2048u << 10));        // 16 KB
    float* pScore    = (float*)(ws + (2064u << 10));        // 256 KB (2 x N)
    int*   pIdx      = (int*)  (ws + (2320u << 10));        // 256 KB (2 x N)
    int*   bestIdx   = (int*)  (ws + (2576u << 10));        // 128 KB
    int*   counts    = (int*)  (ws + (2704u << 10));        // 16 KB
    float* partials  = (float*)(ws + (2720u << 10));        // 128 KB

    hipMemsetAsync(counts, 0, K_EMB * sizeof(int), stream);

    vq_prep    <<<K_EMB,        256, 0, stream>>>(W, Wh, wnorm);
    vq_argmin  <<<128 * KSPLIT, 256, 0, stream>>>(X, Wh, wnorm, pScore, pIdx);
    vq_merge   <<<N_IN / 256,   256, 0, stream>>>(pScore, pIdx, bestIdx);
    vq_gather  <<<N_IN,         256, 0, stream>>>(X, W, bestIdx, out, partials);
    vq_counts  <<<N_IN / 256,   256, 0, stream>>>(bestIdx, counts);
    vq_finalize<<<1,            256, 0, stream>>>(partials, counts, out);
}
